// GRU_52596169507028
// MI455X (gfx1250) — matrix operand order repaired and checked by reading, not compiled, not run on hardware
//
#include <hip/hip_runtime.h>

#define B_ 64
#define T_ 512
#define I_ 1024
#define U_ 1024
#define K_ 2048            // I + U
#define XPAD 8             // bf16 elems of row padding in LDS to spread banks
#define LN_EPS 1e-3f
#define UPD_BIAS_ (-1.0f)
#define NUM_WG 128
#define KC_ 64             // K_/32 k-chunks
#define TILESZ ((size_t)KC_ * 32 * 16)   // halves per packed 16-col B tile

typedef __attribute__((ext_vector_type(16))) __bf16 v16bf;
typedef __attribute__((ext_vector_type(8)))  float  v8f;
typedef __attribute__((ext_vector_type(4)))  unsigned int u32x4;
typedef __attribute__((ext_vector_type(4)))  float  f32x4;

union ABf { u32x4 q[2]; v16bf v; };
union H8  { u32x4 q; __bf16 h[8]; };

// ---------------- init: zero barrier state, pack W fp32 -> bf16 WMMA-B tiles ----
// Packed layout: idx = ((tile*64 + kc)*32 + lane)*16 + e
//   tile = g*64 + j (192 tiles of 16 cols), col = tile*16 + (lane&15),
//   k = kc*32 + (lane>>4)*16 + e  -> lane-contiguous 32B, fully coalesced loads.
__global__ void gru_init_kernel(const float* __restrict__ W,
                                __bf16* __restrict__ wpack,
                                int* __restrict__ bar) {
  if (blockIdx.x == 0 && threadIdx.x < 2) bar[threadIdx.x] = 0;
  const long long n = 3LL * U_ * K_;
  const long long stride = (long long)gridDim.x * blockDim.x;
  for (long long i = (long long)blockIdx.x * blockDim.x + threadIdx.x; i < n; i += stride) {
    const int e    = (int)(i & 15);
    const int lane = (int)((i >> 4) & 31);
    const int kc   = (int)((i >> 9) & 63);
    const long long tile = i >> 15;                    // 0..191
    const int col = (int)(tile * 16) + (lane & 15);    // g*1024 + j*16 + nl
    const int k   = kc * 32 + (lane >> 4) * 16 + e;
    wpack[i] = (__bf16)W[(size_t)col * K_ + k];
  }
}

// ---------------- grid-wide generation barrier (agent scope) ----------------
__device__ __forceinline__ void grid_barrier(int* cnt, int* gen, int target) {
  __syncthreads();
  if (threadIdx.x == 0) {
    int prev = __hip_atomic_fetch_add(cnt, 1, __ATOMIC_ACQ_REL, __HIP_MEMORY_SCOPE_AGENT);
    if (prev == NUM_WG - 1) {
      __hip_atomic_store(cnt, 0, __ATOMIC_RELAXED, __HIP_MEMORY_SCOPE_AGENT);
      __hip_atomic_store(gen, target, __ATOMIC_RELEASE, __HIP_MEMORY_SCOPE_AGENT);
    } else {
      while (__hip_atomic_load(gen, __ATOMIC_ACQUIRE, __HIP_MEMORY_SCOPE_AGENT) < target)
        __builtin_amdgcn_s_sleep(2);
    }
  }
  __syncthreads();
}

__device__ __forceinline__ float fast_rcp(float x) { return __builtin_amdgcn_rcpf(x); }
__device__ __forceinline__ float sigmoidf_(float x) {
  return fast_rcp(1.0f + __expf(-x));                  // v_exp + v_rcp, no div chain
}
__device__ __forceinline__ float tanhf_(float z) {
  z = fminf(fmaxf(z, -20.0f), 20.0f);                  // avoid inf*0
  const float e2 = __expf(2.0f * z);
  return (e2 - 1.0f) * fast_rcp(e2 + 1.0f);
}

// ---------------- persistent scan kernel ----------------
// grid = 128 WGs x 64 threads (2 waves). WG (h = blk>>6, j = blk&63):
//   rows [32h, 32h+32), units [16j, 16j+16). Wave wv owns M-tile rows 16wv..+16.
__global__ void __launch_bounds__(64)
gru_scan_kernel(const float* __restrict__ carry0,
                const float* __restrict__ inputs,         // (B,T,I)
                const unsigned char* __restrict__ resets, // (B,T) bool (1 byte)
                const float* __restrict__ bias,           // (3U)
                const float* __restrict__ ln_s,
                const float* __restrict__ ln_b,
                const __bf16* __restrict__ wpack,         // packed B tiles
                float* __restrict__ out,                  // [B*U final | B*T*U outs]
                int* __restrict__ bar) {
  extern __shared__ char smem_raw[];
  float*  s_lns = (float*)smem_raw;            // K floats
  float*  s_lnb = s_lns + K_;                  // K floats
  __bf16* s_x   = (__bf16*)(s_lnb + K_);       // 32 rows x (K+XPAD) bf16
  const int ROWSTR = K_ + XPAD;

  const int j    = blockIdx.x & 63;
  const int h    = blockIdx.x >> 6;
  const int wv   = threadIdx.x >> 5;
  const int lane = threadIdx.x & 31;

  // stage LN params once
  for (int k = threadIdx.x; k < K_; k += 64) { s_lns[k] = ln_s[k]; s_lnb[k] = ln_b[k]; }
  __syncthreads();

  const int nl = lane & 15;        // N column within tile / M row within tile
  const int hi = lane >> 4;        // lane half
  const int u  = j * 16 + nl;      // unit column this lane produces

  const float br = bias[u];
  const float bc = bias[U_ + u];
  const float bu = bias[2 * U_ + u];

  // packed, lane-contiguous B pointers (per-iter stride = 512 halves)
  const __bf16* bp0 = wpack + (size_t)(0 * 64 + j) * TILESZ + lane * 16;
  const __bf16* bp1 = wpack + (size_t)(1 * 64 + j) * TILESZ + lane * 16;
  const __bf16* bp2 = wpack + (size_t)(2 * 64 + j) * TILESZ + lane * 16;

  float* outs = out + B_ * U_;     // (B,T,U) region
  const int kbase = lane * 64;     // this lane's contiguous K chunk for LN

  for (int t = 0; t < T_; ++t) {
    // carry source: initial carry at t=0, else previous step's output slice
    const float* cprev = (t == 0) ? carry0 : (outs + (size_t)(t - 1) * U_);
    const size_t cstr  = (t == 0) ? (size_t)U_ : (size_t)T_ * U_;

    // ---- Phase A: LayerNorm(concat(c,inp)) -> bf16 x in LDS (wave-local rows) ----
    for (int r = 0; r < 16; ++r) {
      const int rb = 32 * h + 16 * wv + r;                 // batch row
      const bool rst = resets[(size_t)rb * T_ + t] != 0;
      f32x4 vb[16];
      if (lane < 16) {                                     // carry part, k in [0,1024)
        if (rst) {
#pragma unroll
          for (int i = 0; i < 16; ++i) vb[i] = (f32x4)0.0f;
        } else {
          const f32x4* p = (const f32x4*)(cprev + (size_t)rb * cstr + kbase);
#pragma unroll
          for (int i = 0; i < 16; ++i) vb[i] = p[i];
        }
      } else {                                             // input part, k in [1024,2048)
        const f32x4* p = (const f32x4*)(inputs + ((size_t)rb * T_ + t) * I_ + (kbase - I_));
#pragma unroll
        for (int i = 0; i < 16; ++i) vb[i] = p[i];
        if (t + 1 < T_) __builtin_prefetch(p + (I_ / 4), 0, 1);  // next step's input row
      }
      float s = 0.f, s2 = 0.f;
#pragma unroll
      for (int i = 0; i < 16; ++i)
#pragma unroll
        for (int c = 0; c < 4; ++c) { const float x = vb[i][c]; s += x; s2 += x * x; }
#pragma unroll
      for (int m = 16; m >= 1; m >>= 1) {
        s  += __shfl_xor(s,  m, 32);
        s2 += __shfl_xor(s2, m, 32);
      }
      const float mu   = s * (1.0f / (float)K_);
      const float var  = s2 * (1.0f / (float)K_) - mu * mu;
      const float rstd = rsqrtf(var + LN_EPS);

      __bf16* xrow = s_x + (size_t)(16 * wv + r) * ROWSTR + kbase;
#pragma unroll
      for (int i = 0; i < 16; i += 2) {
        H8 hh;
#pragma unroll
        for (int c = 0; c < 8; ++c) {
          const int ii = i + (c >> 2), cc = c & 3;
          const int k  = kbase + i * 4 + c;
          hh.h[c] = (__bf16)((vb[ii][cc] - mu) * rstd * s_lns[k] + s_lnb[k]);
        }
        *(u32x4*)(xrow + i * 4) = hh.q;
      }
    }
    __syncthreads();

    // ---- Phase B: y = x @ W^T for this WG's 3 gate tiles (WMMA bf16) ----
    v8f accR = {br, br, br, br, br, br, br, br};
    v8f accC = {bc, bc, bc, bc, bc, bc, bc, bc};
    v8f accU = {bu, bu, bu, bu, bu, bu, bu, bu};

    // A operand: lane row = 16wv+nl; halves hold k {0..15} / {16..31}
    const __bf16* xa = s_x + (size_t)(16 * wv + nl) * ROWSTR + hi * 16;
#pragma unroll 4
    for (int kc = 0; kc < KC_; ++kc) {
      const int k0 = kc * 32;
      const int bo = kc * 512;                 // halves into packed tile
      ABf a, w0, w1, w2;
      a.q[0]  = *(const u32x4*)(xa + k0);
      a.q[1]  = *(const u32x4*)(xa + k0 + 8);
      w0.q[0] = *(const u32x4*)(bp0 + bo);
      w0.q[1] = *(const u32x4*)(bp0 + bo + 8);
      w1.q[0] = *(const u32x4*)(bp1 + bo);
      w1.q[1] = *(const u32x4*)(bp1 + bo + 8);
      w2.q[0] = *(const u32x4*)(bp2 + bo);
      w2.q[1] = *(const u32x4*)(bp2 + bo + 8);
      accR = __builtin_amdgcn_wmma_f32_16x16x32_bf16(false, a.v, false, w0.v, (short)0, accR, false, false);
      accC = __builtin_amdgcn_wmma_f32_16x16x32_bf16(false, a.v, false, w1.v, (short)0, accC, false, false);
      accU = __builtin_amdgcn_wmma_f32_16x16x32_bf16(false, a.v, false, w2.v, (short)0, accU, false, false);
    }

    // ---- Phase C: gates + carry update (C/D layout: m = v + 8*hi, n = nl) ----
#pragma unroll
    for (int v = 0; v < 8; ++v) {
      const int row = 32 * h + 16 * wv + 8 * hi + v;
      const float sr = sigmoidf_(accR[v]);
      const float cd = tanhf_(sr * accC[v]);
      const float su = sigmoidf_(accU[v] + UPD_BIAS_);
      const bool rst = resets[(size_t)row * T_ + t] != 0;
      const float cold = rst ? 0.0f : cprev[(size_t)row * cstr + u];
      const float cn = su * cd + (1.0f - su) * cold;
      outs[((size_t)row * T_ + t) * U_ + u] = cn;
      if (t == T_ - 1) out[(size_t)row * U_ + u] = cn;   // final carry
    }

    grid_barrier(bar, bar + 1, t + 1);
  }
}

extern "C" void kernel_launch(void* const* d_in, const int* in_sizes, int n_in,
                              void* d_out, int out_size, void* d_ws, size_t ws_size,
                              hipStream_t stream) {
  const float*         carry  = (const float*)d_in[0];
  const float*         inputs = (const float*)d_in[1];
  const unsigned char* resets = (const unsigned char*)d_in[2]; // jax bool = 1 byte
  const float*         W      = (const float*)d_in[3];
  const float*         bias   = (const float*)d_in[4];
  const float*         lns    = (const float*)d_in[5];
  const float*         lnb    = (const float*)d_in[6];
  float*               out    = (float*)d_out;

  int*    bar   = (int*)d_ws;                     // 2 ints of barrier state
  __bf16* wpack = (__bf16*)((char*)d_ws + 256);   // 3U*K bf16 = 12.6 MB (packed tiles)

  gru_init_kernel<<<2048, 256, 0, stream>>>(W, wpack, bar);

  const size_t shmem = (size_t)2 * K_ * sizeof(float)             // LN params
                     + (size_t)32 * (K_ + XPAD) * sizeof(__bf16); // x tile
  (void)hipFuncSetAttribute((const void*)gru_scan_kernel,
                            hipFuncAttributeMaxDynamicSharedMemorySize, (int)shmem);
  gru_scan_kernel<<<NUM_WG, 64, shmem, stream>>>(carry, inputs, resets, bias, lns, lnb,
                                                 wpack, out, bar);
}
